// UniMP_75771813036064
// MI455X (gfx1250) — compile-verified, matrix operand order checked by reading, not run on hardware
//
#include <hip/hip_runtime.h>

typedef __bf16 bf16_t;
typedef __attribute__((ext_vector_type(16))) __bf16 v16bf;
typedef __attribute__((ext_vector_type(8)))  __bf16 v8bf;
typedef __attribute__((ext_vector_type(8)))  float  v8f;

#define HID    128
#define HEADS  8
#define CDIM   16
#define GRAPHS 64
#define OUTD   64
#define EIN    32
#define ENC_NEG_INF 0x007FFFFFu   // encoded -inf for monotone-uint float max

__device__ __forceinline__ unsigned enc_f(float f) {
  unsigned b = __float_as_uint(f);
  return (b & 0x80000000u) ? ~b : (b | 0x80000000u);
}
__device__ __forceinline__ float dec_f(unsigned e) {
  return (e & 0x80000000u) ? __uint_as_float(e & 0x7FFFFFFFu)
                           : __uint_as_float(~e);
}

// ---------------- fills ----------------
__global__ void fill_f32(float* __restrict__ p, float v, long n) {
  long t = (long)blockIdx.x * blockDim.x + threadIdx.x;
  if (t < n) p[t] = v;
}
__global__ void fill_u32(unsigned* __restrict__ p, unsigned v, long n) {
  long t = (long)blockIdx.x * blockDim.x + threadIdx.x;
  if (t < n) p[t] = v;
}

// -------- weight prep: Wt[n*K+k] = (bf16)W[k*N+n] --------
__global__ void prep_wt(const float* __restrict__ W, bf16_t* __restrict__ Wt,
                        int K, int Ncols) {
  int t = blockIdx.x * blockDim.x + threadIdx.x;
  if (t >= K * Ncols) return;
  int n = t / K, k = t - n * K;
  Wt[t] = (bf16_t)W[(size_t)k * Ncols + n];
}

// -------- WMMA operand packing (wave32 16x16x32 bf16 layouts) --------
// A 16x32: lanes 0-15 hold row(lane), K {0..7,16..23}; lanes 16-31 hold K {8..15,24..31}
__device__ __forceinline__ v16bf pack_a_f32(const float* base, int lda, int k0, int lane) {
  int row = lane & 15;
  int ko  = k0 + ((lane & 16) ? 8 : 0);
  const float4* p = (const float4*)(base + (size_t)row * lda + ko);
  float4 u0 = p[0], u1 = p[1], u2 = p[4], u3 = p[5];
  v16bf a;
  a[0]=(bf16_t)u0.x; a[1]=(bf16_t)u0.y; a[2]=(bf16_t)u0.z; a[3]=(bf16_t)u0.w;
  a[4]=(bf16_t)u1.x; a[5]=(bf16_t)u1.y; a[6]=(bf16_t)u1.z; a[7]=(bf16_t)u1.w;
  a[8]=(bf16_t)u2.x; a[9]=(bf16_t)u2.y; a[10]=(bf16_t)u2.z; a[11]=(bf16_t)u2.w;
  a[12]=(bf16_t)u3.x; a[13]=(bf16_t)u3.y; a[14]=(bf16_t)u3.z; a[15]=(bf16_t)u3.w;
  return a;
}
// B 32x16 packed from LDS-staged transposed bf16 weights Wt[N][K]
__device__ __forceinline__ v16bf pack_b_lds(const bf16_t* Wt, int ldb, int n0, int k0, int lane) {
  int n  = n0 + (lane & 15);
  int ko = k0 + ((lane & 16) ? 8 : 0);
  const v8bf* p = (const v8bf*)(Wt + (size_t)n * ldb + ko);
  v8bf lo = p[0], hi = p[2];
  v16bf b;
#pragma unroll
  for (int i = 0; i < 8; i++) { b[i] = lo[i]; b[8 + i] = hi[i]; }
  return b;
}

// -------- GEMM: C[M x NT*16] = act(A[MxK] @ W[KxNT*16] + bias) --------
// 256 threads = 8 waves. Weights staged in LDS once per block; each wave
// computes TWO 16-row tiles so every ds-fetched B feeds two WMMAs. M % 16 == 0.
template <int NT>
__global__ void __launch_bounds__(256)
gemm_wmma(const float* __restrict__ A, const bf16_t* __restrict__ Wt,
          const float* __restrict__ bias, float* __restrict__ C,
          int M, int K, int relu) {
  __shared__ bf16_t smem[NT * 16 * 128];
  const int tid = threadIdx.x;
  // cooperative 16B-chunk weight staging (Wt is 256B aligned, NT*16*K % 2048 == 0)
  {
    int total = NT * 16 * K;
    for (int idx = tid * 8; idx < total; idx += 256 * 8)
      *(uint4*)(smem + idx) = *(const uint4*)(Wt + idx);
  }
  __syncthreads();

  const int wave = tid >> 5;
  const int lane = tid & 31;
  const int row0 = ((blockIdx.x * 8 + wave) * 2) << 4;   // 32 rows per wave
  if (row0 >= M) return;
  const bool two = (row0 + 16) < M;
  const float* A0 = A + (size_t)row0 * K;
  const float* A1 = A + (size_t)(two ? row0 + 16 : row0) * K;

  v8f acc0[NT], acc1[NT];
#pragma unroll
  for (int n = 0; n < NT; n++)
#pragma unroll
    for (int j = 0; j < 8; j++) { acc0[n][j] = 0.0f; acc1[n][j] = 0.0f; }

  const int nks = K >> 5;
  for (int ks = 0; ks < nks; ks++) {
    v16bf a0 = pack_a_f32(A0, K, ks * 32, lane);
    v16bf a1 = pack_a_f32(A1, K, ks * 32, lane);
    if (ks + 1 < nks) {
      __builtin_prefetch(A0 + (size_t)(lane & 15) * K + (ks + 1) * 32, 0, 1);
      __builtin_prefetch(A1 + (size_t)(lane & 15) * K + (ks + 1) * 32, 0, 1);
    }
#pragma unroll
    for (int nt = 0; nt < NT; nt++) {
      v16bf b = pack_b_lds(smem, K, nt * 16, ks * 32, lane);
      acc0[nt] = __builtin_amdgcn_wmma_f32_16x16x32_bf16(
          false, a0, false, b, (short)0, acc0[nt], false, false);
      acc1[nt] = __builtin_amdgcn_wmma_f32_16x16x32_bf16(
          false, a1, false, b, (short)0, acc1[nt], false, false);
    }
  }
  // D layout: VGPR i, lane l -> row = i + (l<16?0:8), col = l&15
  const int colb = lane & 15;
  const int radd = (lane & 16) ? 8 : 0;
  const int ldc = NT * 16;
#pragma unroll
  for (int nt = 0; nt < NT; nt++) {
    int col = nt * 16 + colb;
    float bvv = bias ? bias[col] : 0.0f;
#pragma unroll
    for (int i = 0; i < 8; i++) {
      float v0 = acc0[nt][i] + bvv;
      if (relu) v0 = fmaxf(v0, 0.0f);
      C[(size_t)(row0 + radd + i) * ldc + col] = v0;
      if (two) {
        float v1 = acc1[nt][i] + bvv;
        if (relu) v1 = fmaxf(v1, 0.0f);
        C[(size_t)(row0 + 16 + radd + i) * ldc + col] = v1;
      }
    }
  }
}

// -------- attention: alpha + segment max --------
__global__ void attn_alpha(const float* __restrict__ q, const float* __restrict__ k,
                           const float* __restrict__ e, const int* __restrict__ src,
                           const int* __restrict__ dst, float* __restrict__ alpha,
                           unsigned* __restrict__ amax, int E_) {
  int t = blockIdx.x * blockDim.x + threadIdx.x;
  if (t >= E_ * HEADS) return;
  int eid = t >> 3, h = t & 7;
  int s = src[eid], d = dst[eid];
  const float4* qp = (const float4*)(q + (size_t)d * HID + h * CDIM);
  const float4* kp = (const float4*)(k + (size_t)s * HID + h * CDIM);
  const float4* ep = (const float4*)(e + (size_t)eid * HID + h * CDIM);
  float sum = 0.0f;
#pragma unroll
  for (int i = 0; i < 4; i++) {
    float4 qv = qp[i], kv = kp[i], ev = ep[i];
    sum += qv.x * (kv.x + ev.x) + qv.y * (kv.y + ev.y) +
           qv.z * (kv.z + ev.z) + qv.w * (kv.w + ev.w);
  }
  sum *= 0.25f;  // 1/sqrt(16)
  alpha[t] = sum;
  atomicMax(amax + (size_t)d * HEADS + h, enc_f(sum));
}

__global__ void attn_expsum(float* __restrict__ alpha, const unsigned* __restrict__ amax,
                            float* __restrict__ denom, const int* __restrict__ dst, int E_) {
  int t = blockIdx.x * blockDim.x + threadIdx.x;
  if (t >= E_ * HEADS) return;
  int eid = t >> 3, h = t & 7;
  int d = dst[eid];
  float m = dec_f(amax[(size_t)d * HEADS + h]);
  float ex = __expf(alpha[t] - m);
  alpha[t] = ex;
  atomicAdd(denom + (size_t)d * HEADS + h, ex);
}

__global__ void attn_scatter(const float* __restrict__ alpha, const float* __restrict__ denom,
                             const float* __restrict__ vsrc, const float* __restrict__ e,
                             const int* __restrict__ src, const int* __restrict__ dst,
                             float* __restrict__ outagg, int E_) {
  int t = blockIdx.x * blockDim.x + threadIdx.x;
  if (t >= E_ * HEADS) return;
  int eid = t >> 3, h = t & 7;
  int s = src[eid], d = dst[eid];
  float a = alpha[t] / denom[(size_t)d * HEADS + h];
  const float4* vp = (const float4*)(vsrc + (size_t)s * HID + h * CDIM);
  const float4* ep = (const float4*)(e + (size_t)eid * HID + h * CDIM);
  float* op = outagg + (size_t)d * HID + h * CDIM;
#pragma unroll
  for (int i = 0; i < 4; i++) {
    float4 vq = vp[i], eq = ep[i];
    atomicAdd(op + 4 * i + 0, (vq.x + eq.x) * a);
    atomicAdd(op + 4 * i + 1, (vq.y + eq.y) * a);
    atomicAdd(op + 4 * i + 2, (vq.z + eq.z) * a);
    atomicAdd(op + 4 * i + 3, (vq.w + eq.w) * a);
  }
}

// -------- beta-gated skip: one block (128 thr) per node --------
__global__ void beta_gate(const float* __restrict__ outagg, const float* __restrict__ r,
                          const float* __restrict__ wbeta, float* __restrict__ x, int N_) {
  int n = blockIdx.x;
  int c = threadIdx.x;
  float o  = outagg[(size_t)n * HID + c];
  float rr = r[(size_t)n * HID + c];
  float part = o * wbeta[c] + rr * wbeta[HID + c] + (o - rr) * wbeta[2 * HID + c];
  __shared__ float red[HID];
  red[c] = part;
  __syncthreads();
  for (int s = 64; s > 0; s >>= 1) {
    if (c < s) red[c] += red[c + s];
    __syncthreads();
  }
  float beta = 1.0f / (1.0f + __expf(-red[0]));
  x[(size_t)n * HID + c] = beta * rr + (1.0f - beta) * o;
}

// -------- BatchNorm (training-mode stats over nodes) --------
__global__ void bn_stats(const float* __restrict__ x, float* __restrict__ acc, int N_) {
  int c = threadIdx.x;
  float s = 0.0f, sq = 0.0f;
  for (long n = blockIdx.x; n < N_; n += gridDim.x) {
    float v = x[n * HID + c];
    s += v; sq += v * v;
  }
  atomicAdd(acc + c, s);
  atomicAdd(acc + HID + c, sq);
}
__global__ void bn_apply(float* __restrict__ x, const float* __restrict__ acc,
                         const float* __restrict__ gamma, const float* __restrict__ beta,
                         int N_) {
  long t = (long)blockIdx.x * blockDim.x + threadIdx.x;
  if (t >= (long)N_ * HID) return;
  int c = (int)(t & (HID - 1));
  float inv_n = 1.0f / (float)N_;
  float mu = acc[c] * inv_n;
  float var = acc[HID + c] * inv_n - mu * mu;
  x[t] = (x[t] - mu) * rsqrtf(var + 1e-5f) * gamma[c] + beta[c];
}

// -------- TopK pooling with min_score --------
__global__ void pool_score(const float* __restrict__ x, const float* __restrict__ pw,
                           float* __restrict__ s, unsigned* __restrict__ smax,
                           const int* __restrict__ batch, int N_) {
  int n = blockIdx.x * blockDim.x + threadIdx.x;
  if (n >= N_) return;
  const float4* xp = (const float4*)(x + (size_t)n * HID);
  const float4* wp = (const float4*)pw;
  float sum = 0.0f;
#pragma unroll
  for (int i = 0; i < HID / 4; i++) {
    float4 a = xp[i], b = wp[i];
    sum += a.x * b.x + a.y * b.y + a.z * b.z + a.w * b.w;
  }
  s[n] = sum;
  atomicMax(smax + batch[n], enc_f(sum));
}
__global__ void pool_expsum(const float* __restrict__ s, const unsigned* __restrict__ smax,
                            float* __restrict__ es, float* __restrict__ sden,
                            const int* __restrict__ batch, int N_) {
  int n = blockIdx.x * blockDim.x + threadIdx.x;
  if (n >= N_) return;
  int g = batch[n];
  float ex = __expf(s[n] - dec_f(smax[g]));
  es[n] = ex;
  atomicAdd(sden + g, ex);
}
__global__ void pool_agg(const float* __restrict__ x, const float* __restrict__ es,
                         const float* __restrict__ sden, const int* __restrict__ batch,
                         float* __restrict__ pooled, float* __restrict__ cnt, int N_) {
  int n = blockIdx.x;
  int c = threadIdx.x;
  int g = batch[n];
  float score = es[n] / sden[g];
  float keep = (score > 0.001f) ? 1.0f : 0.0f;
  atomicAdd(pooled + (size_t)g * HID + c, x[(size_t)n * HID + c] * score * keep);
  if (c == 0) atomicAdd(cnt + g, keep);
}
__global__ void pool_div(float* __restrict__ pooled, const float* __restrict__ cnt, int G_) {
  int t = blockIdx.x * blockDim.x + threadIdx.x;
  if (t >= G_ * HID) return;
  pooled[t] /= fmaxf(cnt[t >> 7], 1.0f);
}

// ============================================================
extern "C" void kernel_launch(void* const* d_in, const int* in_sizes, int n_in,
                              void* d_out, int out_size, void* d_ws, size_t ws_size,
                              hipStream_t stream) {
  (void)n_in; (void)out_size; (void)ws_size;
  const float* x_in   = (const float*)d_in[0];
  const int*   ei     = (const int*)  d_in[1];
  const float* eattr  = (const float*)d_in[2];
  const int*   batch  = (const int*)  d_in[3];
  const float* node_w1 = (const float*)d_in[4];  const float* node_b1 = (const float*)d_in[5];
  const float* node_w2 = (const float*)d_in[6];  const float* node_b2 = (const float*)d_in[7];
  const float* edge_w1 = (const float*)d_in[8];  const float* edge_b1 = (const float*)d_in[9];
  const float* edge_w2 = (const float*)d_in[10]; const float* edge_b2 = (const float*)d_in[11];
  const float* wq = (const float*)d_in[12]; const float* bq = (const float*)d_in[13];
  const float* wk = (const float*)d_in[14]; const float* bk = (const float*)d_in[15];
  const float* wv = (const float*)d_in[16]; const float* bvv = (const float*)d_in[17];
  const float* we = (const float*)d_in[18];
  const float* wskip = (const float*)d_in[19]; const float* bskip = (const float*)d_in[20];
  const float* wbeta = (const float*)d_in[21];
  const float* bn_g = (const float*)d_in[22]; const float* bn_b = (const float*)d_in[23];
  const float* pin_w1 = (const float*)d_in[24]; const float* pin_b1 = (const float*)d_in[25];
  const float* pin_w2 = (const float*)d_in[26]; const float* pin_b2 = (const float*)d_in[27];
  const float* pool_w = (const float*)d_in[28];
  const float* out_w1 = (const float*)d_in[29]; const float* out_b1 = (const float*)d_in[30];
  const float* out_w2 = (const float*)d_in[31]; const float* out_b2 = (const float*)d_in[32];

  const int N_ = in_sizes[0] / HID;
  const int E_ = in_sizes[1] / 2;
  const int L_ = 2;
  const int* src = ei;
  const int* dst = ei + E_;

  // ---- workspace bump allocator ----
  char* base = (char*)d_ws;
  size_t off = 0;
  auto alloc = [&](size_t bytes) -> void* {
    void* p = base + off;
    off += (bytes + 255) & ~(size_t)255;
    return p;
  };
  float* xbuf   = (float*)alloc((size_t)N_ * HID * 4);
  float* qbuf   = (float*)alloc((size_t)N_ * HID * 4);
  float* kbuf   = (float*)alloc((size_t)N_ * HID * 4);
  float* vbuf   = (float*)alloc((size_t)N_ * HID * 4);
  float* rbuf   = (float*)alloc((size_t)N_ * HID * 4);
  float* outagg = (float*)alloc((size_t)N_ * HID * 4);
  float* eabuf  = (float*)alloc((size_t)E_ * HID * 4);
  float* ebuf   = (float*)alloc((size_t)E_ * HID * 4);
  float* alpha  = (float*)alloc((size_t)E_ * HEADS * 4);
  unsigned* amax = (unsigned*)alloc((size_t)N_ * HEADS * 4);
  float* denom  = (float*)alloc((size_t)N_ * HEADS * 4);
  float* bnacc  = (float*)alloc(2 * HID * 4);
  float* sbuf   = (float*)alloc((size_t)N_ * 4);
  float* esbuf  = (float*)alloc((size_t)N_ * 4);
  unsigned* smax = (unsigned*)alloc(GRAPHS * 4);
  float* sden   = (float*)alloc(GRAPHS * 4);
  float* cnt    = (float*)alloc(GRAPHS * 4);
  float* pooled = (float*)alloc(GRAPHS * HID * 4);
  float* pooled2= (float*)alloc(GRAPHS * HID * 4);
  const size_t WSZ = (size_t)HID * HID;  // 16384 elems
  bf16_t* node_w1t = (bf16_t*)alloc(WSZ * 2);
  bf16_t* node_w2t = (bf16_t*)alloc(WSZ * 2);
  bf16_t* edge_w1t = (bf16_t*)alloc((size_t)EIN * HID * 2);
  bf16_t* edge_w2t = (bf16_t*)alloc(WSZ * 2);
  bf16_t* wq_t = (bf16_t*)alloc(2 * WSZ * 2);
  bf16_t* wk_t = (bf16_t*)alloc(2 * WSZ * 2);
  bf16_t* wv_t = (bf16_t*)alloc(2 * WSZ * 2);
  bf16_t* we_t = (bf16_t*)alloc(2 * WSZ * 2);
  bf16_t* ws_t = (bf16_t*)alloc(2 * WSZ * 2);
  bf16_t* pin_w1t = (bf16_t*)alloc(WSZ * 2);
  bf16_t* pin_w2t = (bf16_t*)alloc(WSZ * 2);
  bf16_t* out_w1t = (bf16_t*)alloc(WSZ * 2);
  bf16_t* out_w2t = (bf16_t*)alloc((size_t)HID * OUTD * 2);

  auto prep = [&](const float* W, bf16_t* Wt, int K, int Ncols) {
    int n = K * Ncols;
    prep_wt<<<dim3((n + 255) / 256), dim3(256), 0, stream>>>(W, Wt, K, Ncols);
  };
  prep(node_w1, node_w1t, HID, HID);
  prep(node_w2, node_w2t, HID, HID);
  prep(edge_w1, edge_w1t, EIN, HID);
  prep(edge_w2, edge_w2t, HID, HID);
  for (int i = 0; i < L_; i++) {
    prep(wq + (size_t)i * WSZ, wq_t + (size_t)i * WSZ, HID, HID);
    prep(wk + (size_t)i * WSZ, wk_t + (size_t)i * WSZ, HID, HID);
    prep(wv + (size_t)i * WSZ, wv_t + (size_t)i * WSZ, HID, HID);
    prep(we + (size_t)i * WSZ, we_t + (size_t)i * WSZ, HID, HID);
    prep(wskip + (size_t)i * WSZ, ws_t + (size_t)i * WSZ, HID, HID);
  }
  prep(pin_w1, pin_w1t, HID, HID);
  prep(pin_w2, pin_w2t, HID, HID);
  prep(out_w1, out_w1t, HID, HID);
  prep(out_w2, out_w2t, HID, OUTD);

  auto gemm8 = [&](const float* A, const bf16_t* Wt, const float* bias,
                   float* C, int M, int K, int relu) {
    gemm_wmma<8><<<dim3((M + 255) / 256), dim3(256), 0, stream>>>(A, Wt, bias, C, M, K, relu);
  };
  auto gemm4 = [&](const float* A, const bf16_t* Wt, const float* bias,
                   float* C, int M, int K, int relu) {
    gemm_wmma<4><<<dim3((M + 255) / 256), dim3(256), 0, stream>>>(A, Wt, bias, C, M, K, relu);
  };
  auto zf = [&](float* p, long n) {
    fill_f32<<<dim3((unsigned)((n + 255) / 256)), dim3(256), 0, stream>>>(p, 0.0f, n);
  };
  auto zu = [&](unsigned* p, unsigned v, long n) {
    fill_u32<<<dim3((unsigned)((n + 255) / 256)), dim3(256), 0, stream>>>(p, v, n);
  };

  // ---- node MLP, edge MLP ----
  gemm8(x_in, node_w1t, node_b1, qbuf, N_, HID, 1);
  gemm8(qbuf, node_w2t, node_b2, xbuf, N_, HID, 0);
  gemm8(eattr, edge_w1t, edge_b1, ebuf, E_, EIN, 1);
  gemm8(ebuf, edge_w2t, edge_b2, eabuf, E_, HID, 0);

  const int EH_BLOCKS = (E_ * HEADS + 255) / 256;
  for (int i = 0; i < L_; i++) {
    const bf16_t* wqt = wq_t + (size_t)i * WSZ;
    const bf16_t* wkt = wk_t + (size_t)i * WSZ;
    const bf16_t* wvt = wv_t + (size_t)i * WSZ;
    const bf16_t* wet = we_t + (size_t)i * WSZ;
    const bf16_t* wst = ws_t + (size_t)i * WSZ;
    gemm8(xbuf, wqt, bq + (size_t)i * HID, qbuf, N_, HID, 0);
    gemm8(xbuf, wkt, bk + (size_t)i * HID, kbuf, N_, HID, 0);
    gemm8(xbuf, wvt, bvv + (size_t)i * HID, vbuf, N_, HID, 0);
    gemm8(eabuf, wet, nullptr, ebuf, E_, HID, 0);

    zu(amax, ENC_NEG_INF, (long)N_ * HEADS);
    zf(denom, (long)N_ * HEADS);
    attn_alpha<<<dim3(EH_BLOCKS), dim3(256), 0, stream>>>(qbuf, kbuf, ebuf, src, dst, alpha, amax, E_);
    attn_expsum<<<dim3(EH_BLOCKS), dim3(256), 0, stream>>>(alpha, amax, denom, dst, E_);
    zf(outagg, (long)N_ * HID);
    attn_scatter<<<dim3(EH_BLOCKS), dim3(256), 0, stream>>>(alpha, denom, vbuf, ebuf, src, dst, outagg, E_);

    gemm8(xbuf, wst, bskip + (size_t)i * HID, rbuf, N_, HID, 0);
    beta_gate<<<dim3(N_), dim3(HID), 0, stream>>>(outagg, rbuf, wbeta + (size_t)i * 3 * HID, xbuf, N_);

    zf(bnacc, 2 * HID);
    bn_stats<<<dim3(480), dim3(HID), 0, stream>>>(xbuf, bnacc, N_);
    bn_apply<<<dim3((unsigned)(((long)N_ * HID + 255) / 256)), dim3(256), 0, stream>>>(
        xbuf, bnacc, bn_g + (size_t)i * HID, bn_b + (size_t)i * HID, N_);
  }

  // ---- pool_in MLP ----
  gemm8(xbuf, pin_w1t, pin_b1, qbuf, N_, HID, 1);
  gemm8(qbuf, pin_w2t, pin_b2, rbuf, N_, HID, 0);

  // ---- TopK pooling ----
  zu(smax, ENC_NEG_INF, GRAPHS);
  zf(sden, GRAPHS);
  zf(cnt, GRAPHS);
  zf(pooled, GRAPHS * HID);
  pool_score<<<dim3((N_ + 255) / 256), dim3(256), 0, stream>>>(rbuf, pool_w, sbuf, smax, batch, N_);
  pool_expsum<<<dim3((N_ + 255) / 256), dim3(256), 0, stream>>>(sbuf, smax, esbuf, sden, batch, N_);
  pool_agg<<<dim3(N_), dim3(HID), 0, stream>>>(rbuf, esbuf, sden, batch, pooled, cnt, N_);
  pool_div<<<dim3((GRAPHS * HID + 255) / 256), dim3(256), 0, stream>>>(pooled, cnt, GRAPHS);

  // ---- output MLP ----
  gemm8(pooled, out_w1t, out_b1, pooled2, GRAPHS, HID, 1);
  gemm4(pooled2, out_w2t, out_b2, (float*)d_out, GRAPHS, HID, 0);
}
